// LDAGNN_901943132286
// MI455X (gfx1250) — matrix operand order repaired and checked by reading, not compiled, not run on hardware
//
#include <hip/hip_runtime.h>
#include <cstdint>
#include <cstddef>

// ---------------------------------------------------------------------------
// LDAGNN on MI455X (gfx1250, wave32). All dense GEMMs use
// v_wmma_f32_16x16x32_bf16 (f32 accumulate). Attention matrices are converted
// to bf16 on the fly; 4096^2 E-matrix fits in the 192MB L2 so the big GEMMs
// are matrix-unit bound -> bf16 WMMA (K=32/instr) is the right precision path.
// ---------------------------------------------------------------------------

#define EMB   768
#define NHEAD 4
#define N0C   2048
#define N1C   4096
#define BATCH 32
#define DOCT  200
#define NCLS  20

typedef __attribute__((ext_vector_type(16))) __bf16 v16bf;
typedef __attribute__((ext_vector_type(8)))  float  v8f;

union ABFrag { uint4 u[2]; v16bf v; };

__device__ __forceinline__ unsigned short f2bf(float f) {
  unsigned int u = __float_as_uint(f);
  u += 0x7fffu + ((u >> 16) & 1u);          // round-to-nearest-even
  return (unsigned short)(u >> 16);
}

// ---------------------------------------------------------------------------
// WMMA bf16 GEMM:  C[M,N] = act( A[M,K](bf16,row) * Bt[N,K](bf16,row)^T )
// Block tile 128x128x32, 8 waves (2 along M x 4 along N), wave tile 64x32.
// act: 0=none, 1=elu, 2=relu.  All dims are multiples of the tile by design.
// ---------------------------------------------------------------------------
__global__ __launch_bounds__(256) void gemm_bf16_kernel(
    const unsigned short* __restrict__ A, const unsigned short* __restrict__ Bt,
    float* __restrict__ C, int M, int Nn, int K,
    int lda, int ldb, int ldc, int act)
{
  (void)M; (void)Nn;
  __shared__ unsigned short As[128 * 40];   // 32 bf16 + 8 pad -> 80B rows (16B aligned)
  __shared__ unsigned short Bs[128 * 40];

  const int tid  = threadIdx.x;
  const int lane = tid & 31;
  const int wid  = tid >> 5;
  const int wm   = wid & 1;                 // 2 waves along M
  const int wn   = wid >> 1;                // 4 waves along N
  const int m0   = blockIdx.y * 128;
  const int n0   = blockIdx.x * 128;

  v8f acc[4][2];
  v8f zero = {};
#pragma unroll
  for (int m = 0; m < 4; ++m)
#pragma unroll
    for (int n = 0; n < 2; ++n) acc[m][n] = zero;

  const int klA = (lane & 16) ? 8 : 0;      // A frag: K elems klA..+7 and klA+16..+23
  const int klB = (lane & 16) ? 8 : 0;      // B frag: K elems klB..+7 and klB+16..+23
  const int rsub = lane & 15;

  for (int k0 = 0; k0 < K; k0 += 32) {
    // Stage 128x32 bf16 tiles of A and Bt: 512 16B chunks / 256 threads = 2 each.
#pragma unroll
    for (int i = 0; i < 2; ++i) {
      int chunk = tid + i * 256;
      int r  = chunk >> 2;
      int c8 = (chunk & 3) << 3;
      uint4 va = *(const uint4*)(A  + (size_t)(m0 + r) * lda + k0 + c8);
      *(uint4*)(As + r * 40 + c8) = va;
      uint4 vb = *(const uint4*)(Bt + (size_t)(n0 + r) * ldb + k0 + c8);
      *(uint4*)(Bs + r * 40 + c8) = vb;
    }
    __syncthreads();

    ABFrag fa[4], fb[2];
#pragma unroll
    for (int m = 0; m < 4; ++m) {
      const unsigned short* ap = As + (wm * 64 + m * 16 + rsub) * 40;
      fa[m].u[0] = *(const uint4*)(ap + klA);
      fa[m].u[1] = *(const uint4*)(ap + klA + 16);
    }
#pragma unroll
    for (int n = 0; n < 2; ++n) {
      const unsigned short* bp = Bs + (wn * 32 + n * 16 + rsub) * 40;
      fb[n].u[0] = *(const uint4*)(bp + klB);
      fb[n].u[1] = *(const uint4*)(bp + klB + 16);
    }

#pragma unroll
    for (int m = 0; m < 4; ++m)
#pragma unroll
      for (int n = 0; n < 2; ++n)
        acc[m][n] = __builtin_amdgcn_wmma_f32_16x16x32_bf16(
            false, fa[m].v, false, fb[n].v, (short)0, acc[m][n], false, false);
    __syncthreads();
  }

  // Epilogue: C 16x16 layout -> lane col = lane&15, VGPR r -> row r + (lane&16?8:0)
  const int rsel = (lane & 16) ? 8 : 0;
#pragma unroll
  for (int m = 0; m < 4; ++m)
#pragma unroll
    for (int n = 0; n < 2; ++n) {
      int col  = n0 + wn * 32 + n * 16 + rsub;
      int rowb = m0 + wm * 64 + m * 16 + rsel;
#pragma unroll
      for (int r = 0; r < 8; ++r) {
        float v = acc[m][n][r];
        if (act == 1)      v = (v > 0.f) ? v : (expf(v) - 1.f);   // elu
        else if (act == 2) v = fmaxf(v, 0.f);                     // relu
        C[(size_t)(rowb + r) * ldc + col] = v;
      }
    }
}

// ---------------------------------------------------------------------------
// f32 -> bf16 flat convert
// ---------------------------------------------------------------------------
__global__ void conv_bf16_kernel(const float* __restrict__ in,
                                 unsigned short* __restrict__ out, size_t n) {
  size_t i = (size_t)blockIdx.x * 256 + threadIdx.x;
  if (i < n) out[i] = f2bf(in[i]);
}

// f32 [R,C] -> bf16 [C,R] transpose-convert (builds GEMM Bt operands)
__global__ void tconv_bf16_kernel(const float* __restrict__ in,
                                  unsigned short* __restrict__ out, int R, int C) {
  size_t i = (size_t)blockIdx.x * 256 + threadIdx.x;
  if (i >= (size_t)R * C) return;
  int r = (int)(i / C), c = (int)(i - (size_t)r * C);
  out[(size_t)c * R + r] = f2bf(in[i]);
}

// ---------------------------------------------------------------------------
// e1[i] = Wh[i,:]·a[0:768] ; e2[i] = Wh[i,:]·a[768:1536]
// ---------------------------------------------------------------------------
__global__ __launch_bounds__(256) void matvec2_kernel(
    const float* __restrict__ Wh, const float* __restrict__ a,
    float* __restrict__ e1v, float* __restrict__ e2v)
{
  __shared__ float red[256];
  int i = blockIdx.x, tid = threadIdx.x;
  const float* row = Wh + (size_t)i * EMB;
  float s1 = 0.f, s2 = 0.f;
  for (int e = tid; e < EMB; e += 256) {
    float w = row[e];
    s1 += w * a[e];
    s2 += w * a[EMB + e];
  }
  red[tid] = s1; __syncthreads();
  for (int st = 128; st > 0; st >>= 1) { if (tid < st) red[tid] += red[tid + st]; __syncthreads(); }
  float t1 = red[0]; __syncthreads();
  red[tid] = s2; __syncthreads();
  for (int st = 128; st > 0; st >>= 1) { if (tid < st) red[tid] += red[tid + st]; __syncthreads(); }
  if (tid == 0) { e1v[i] = t1; e2v[i] = red[0]; }
}

// ---------------------------------------------------------------------------
// Masked row softmax of leaky_relu(e1[i]+e2[j]).
// mode 0: Eout[i,j] = 0.5*adj + 0.5*att        (f32)
// mode 1: mixOut[i,j] = bf16(att * Ein[i,j])   (adaptive-weighted attention)
// ---------------------------------------------------------------------------
__global__ __launch_bounds__(256) void attn_kernel(
    const float* __restrict__ adj, const float* __restrict__ e1v,
    const float* __restrict__ e2v, const float* __restrict__ Ein,
    float* __restrict__ Eout, unsigned short* __restrict__ mixOut,
    int N, int mode)
{
  __shared__ float red[256];
  int i = blockIdx.x, tid = threadIdx.x;
  const float* arow = adj + (size_t)i * N;
  float ei = e1v[i];

  float m = -3.4e38f;
  for (int j = tid; j < N; j += 256) {
    float x = ei + e2v[j];
    x = (x > 0.f) ? x : 0.2f * x;
    float v = (arow[j] > 0.f) ? x : -1e9f;
    m = fmaxf(m, v);
  }
  red[tid] = m; __syncthreads();
  for (int st = 128; st > 0; st >>= 1) { if (tid < st) red[tid] = fmaxf(red[tid], red[tid + st]); __syncthreads(); }
  m = red[0]; __syncthreads();

  float s = 0.f;
  for (int j = tid; j < N; j += 256) {
    float x = ei + e2v[j];
    x = (x > 0.f) ? x : 0.2f * x;
    float v = (arow[j] > 0.f) ? x : -1e9f;
    s += expf(v - m);
  }
  red[tid] = s; __syncthreads();
  for (int st = 128; st > 0; st >>= 1) { if (tid < st) red[tid] += red[tid + st]; __syncthreads(); }
  float inv = 1.f / red[0]; __syncthreads();

  for (int j = tid; j < N; j += 256) {
    float x = ei + e2v[j];
    x = (x > 0.f) ? x : 0.2f * x;
    float v = (arow[j] > 0.f) ? x : -1e9f;
    float att = expf(v - m) * inv;
    size_t o = (size_t)i * N + j;
    if (mode == 0) Eout[o] = 0.5f * arow[j] + 0.5f * att;
    else           mixOut[o] = f2bf(att * Ein[o]);
  }
}

// En bf16[i,j] = d_i * d_j * E[i,j],  d = deg>0 ? deg^-0.5 : 0
__global__ __launch_bounds__(256) void en_conv_kernel(
    const float* __restrict__ Ein, const float* __restrict__ deg,
    unsigned short* __restrict__ out, int N)
{
  int i = blockIdx.x;
  float dgi = deg[i];
  float di = (dgi > 0.f) ? rsqrtf(dgi) : 0.f;
  for (int j = threadIdx.x; j < N; j += 256) {
    float dgj = deg[j];
    float dj = (dgj > 0.f) ? rsqrtf(dgj) : 0.f;
    out[(size_t)i * N + j] = f2bf(di * dj * Ein[(size_t)i * N + j]);
  }
}

// F = 0.5*L + 0.5*Z
__global__ void combine_kernel(const float* __restrict__ L, const float* __restrict__ Z,
                               float* __restrict__ F, int n) {
  int i = blockIdx.x * 256 + threadIdx.x;
  if (i < n) F[i] = 0.5f * L[i] + 0.5f * Z[i];
}

// ---------------------------------------------------------------------------
// Per-node softmax gate over the 4 branch outputs -> R[n,:]
// ---------------------------------------------------------------------------
__global__ __launch_bounds__(256) void fuse_gate_kernel(
    const float* __restrict__ F0, const float* __restrict__ F1,
    const float* __restrict__ F2, const float* __restrict__ F3,
    const float* __restrict__ afuse, float* __restrict__ R)
{
  __shared__ float red[256];
  __shared__ float g[4];
  int n = blockIdx.x, tid = threadIdx.x;
  size_t base = (size_t)n * EMB;
  float s0 = 0.f, s1 = 0.f, s2 = 0.f, s3 = 0.f;
  for (int e = tid; e < EMB; e += 256) {
    float a = afuse[e];
    s0 += F0[base + e] * a; s1 += F1[base + e] * a;
    s2 += F2[base + e] * a; s3 += F3[base + e] * a;
  }
  float sv[4] = { s0, s1, s2, s3 };
  for (int k = 0; k < 4; ++k) {
    red[tid] = sv[k]; __syncthreads();
    for (int st = 128; st > 0; st >>= 1) { if (tid < st) red[tid] += red[tid + st]; __syncthreads(); }
    if (tid == 0) g[k] = red[0];
    __syncthreads();
  }
  if (tid == 0) {
    float m = fmaxf(fmaxf(g[0], g[1]), fmaxf(g[2], g[3]));
    float e0 = expf(g[0] - m), e1 = expf(g[1] - m), e2 = expf(g[2] - m), e3 = expf(g[3] - m);
    float inv = 1.f / (e0 + e1 + e2 + e3);
    g[0] = e0 * inv; g[1] = e1 * inv; g[2] = e2 * inv; g[3] = e3 * inv;
  }
  __syncthreads();
  for (int e = tid; e < EMB; e += 256)
    R[base + e] = g[0] * F0[base + e] + g[1] * F1[base + e] +
                  g[2] * F2[base + e] + g[3] * F3[base + e];
}

// ---------------------------------------------------------------------------
// Fused gather + positional-encoding + mask + classifier matvec + tanh.
// One block per (b,c) pair; FV is never materialized.
// ---------------------------------------------------------------------------
__global__ __launch_bounds__(256) void classify_kernel(
    const float* __restrict__ R, const int* __restrict__ doc_idx,
    const int* __restrict__ doc_lens, const int* __restrict__ subp,
    const float* __restrict__ Wcls, const float* __restrict__ bcls,
    float* __restrict__ logits)
{
  __shared__ float red[256];
  int b = blockIdx.x / NCLS, c = blockIdx.x % NCLS;
  int tid = threadIdx.x;
  int len = doc_lens[b];
  int sub = subp[0];
  const float k_ln = 9.210340371976184f / (float)EMB;   // ln(10000)/768
  float acc = 0.f;
  for (int f = tid; f < DOCT * EMB; f += 256) {
    int t = f / EMB, e = f - t * EMB;
    float x = 0.f;
    if (t < len) {
      int node = doc_idx[b * DOCT + t] + sub;
      float dt  = expf(-(float)(e & ~1) * k_ln);
      float arg = (float)t * dt;
      float pe  = (e & 1) ? cosf(arg) : sinf(arg);
      x = R[(size_t)node * EMB + e] + pe;
    }
    acc += x * Wcls[(size_t)f * NCLS + c];
  }
  red[tid] = acc; __syncthreads();
  for (int st = 128; st > 0; st >>= 1) { if (tid < st) red[tid] += red[tid + st]; __syncthreads(); }
  if (tid == 0) logits[b * NCLS + c] = tanhf(red[0] + bcls[c]);
}

__global__ void softmax20_kernel(const float* __restrict__ logits, float* __restrict__ out) {
  int b = threadIdx.x;
  if (b >= BATCH) return;
  float m = -3.4e38f;
  for (int c = 0; c < NCLS; ++c) m = fmaxf(m, logits[b * NCLS + c]);
  float s = 0.f;
  for (int c = 0; c < NCLS; ++c) s += expf(logits[b * NCLS + c] - m);
  float inv = 1.f / s;
  for (int c = 0; c < NCLS; ++c) out[b * NCLS + c] = expf(logits[b * NCLS + c] - m) * inv;
}

// ---------------------------------------------------------------------------
// Host-side branch driver (all launches on `stream`, graph-capture safe)
// ---------------------------------------------------------------------------
static void run_branch(hipStream_t stream, int N,
                       const unsigned short* hb, const float* adj, const float* deg,
                       const unsigned short* wAttT, const unsigned short* wHeadsT,
                       const unsigned short* wOutT, const unsigned short* wGcnT,
                       const float* a_att, const float* a_heads, const float* a_out,
                       float* wh, unsigned short* wht, float* Ebuf, unsigned short* mixb,
                       float* e1, float* e2, float* xbuf, unsigned short* xb,
                       float* Lbuf, float* Zbuf, float* Fout)
{
  dim3 gg(EMB / 128, N / 128);
  auto gemm = [&](const unsigned short* Ap, const unsigned short* Btp, float* Cp,
                  int K, int lda, int ldb, int ldc, int act) {
    gemm_bf16_kernel<<<gg, 256, 0, stream>>>(Ap, Btp, Cp, N, EMB, K, lda, ldb, ldc, act);
  };
  unsigned int gNE  = (unsigned int)(((size_t)N * EMB + 255) / 256);
  unsigned int gNE4 = (unsigned int)(((size_t)N * NHEAD * EMB + 255) / 256);

  // E = 0.5*adj + 0.5*softmax(leaky_relu(e1+e2))  (attention branch)
  gemm(hb, wAttT, wh, EMB, EMB, EMB, EMB, 0);
  matvec2_kernel<<<N, 256, 0, stream>>>(wh, a_att, e1, e2);
  attn_kernel<<<N, 256, 0, stream>>>(adj, e1, e2, nullptr, Ebuf, nullptr, N, 0);

  // 4 heads -> elu(mix @ Wh) written into concat buffer x[N, 3072]
  for (int k = 0; k < NHEAD; ++k) {
    gemm(hb, wHeadsT + (size_t)k * EMB * EMB, wh, EMB, EMB, EMB, EMB, 0);
    matvec2_kernel<<<N, 256, 0, stream>>>(wh, a_heads + (size_t)k * 2 * EMB, e1, e2);
    attn_kernel<<<N, 256, 0, stream>>>(adj, e1, e2, Ebuf, nullptr, mixb, N, 1);
    tconv_bf16_kernel<<<gNE, 256, 0, stream>>>(wh, wht, N, EMB);
    gemm(mixb, wht, xbuf + (size_t)k * EMB, N, N, N, NHEAD * EMB, 1 /*elu*/);
  }

  // output GAT layer: L = elu(mix_o @ (x @ W_out))
  conv_bf16_kernel<<<gNE4, 256, 0, stream>>>(xbuf, xb, (size_t)N * NHEAD * EMB);
  gemm(xb, wOutT, wh, NHEAD * EMB, NHEAD * EMB, NHEAD * EMB, EMB, 0);
  matvec2_kernel<<<N, 256, 0, stream>>>(wh, a_out, e1, e2);
  attn_kernel<<<N, 256, 0, stream>>>(adj, e1, e2, Ebuf, nullptr, mixb, N, 1);
  tconv_bf16_kernel<<<gNE, 256, 0, stream>>>(wh, wht, N, EMB);
  gemm(mixb, wht, Lbuf, N, N, N, EMB, 1 /*elu*/);

  // GCN branch: Z = relu( (D^-1/2 E D^-1/2) @ (h @ W_gcn) )
  gemm(hb, wGcnT, wh, EMB, EMB, EMB, EMB, 0);
  en_conv_kernel<<<N, 256, 0, stream>>>(Ebuf, deg, mixb, N);
  tconv_bf16_kernel<<<gNE, 256, 0, stream>>>(wh, wht, N, EMB);
  gemm(mixb, wht, Zbuf, N, N, N, EMB, 2 /*relu*/);

  combine_kernel<<<gNE, 256, 0, stream>>>(Lbuf, Zbuf, Fout, N * EMB);
}

extern "C" void kernel_launch(void* const* d_in, const int* in_sizes, int n_in,
                              void* d_out, int out_size, void* d_ws, size_t ws_size,
                              hipStream_t stream) {
  (void)in_sizes; (void)n_in; (void)out_size; (void)ws_size;

  const float* H       = (const float*)d_in[0];
  const float* H_      = (const float*)d_in[1];
  const float* A0      = (const float*)d_in[2];
  const float* A1      = (const float*)d_in[3];
  const float* A2      = (const float*)d_in[4];
  const float* A3      = (const float*)d_in[5];
  const float* Dg      = (const float*)d_in[6];
  const float* Dg_     = (const float*)d_in[7];
  const int*   doc_idx = (const int*)d_in[8];
  const int*   doc_len = (const int*)d_in[9];
  const int*   subp    = (const int*)d_in[10];
  const float* W_att   = (const float*)d_in[11];
  const float* a_att   = (const float*)d_in[12];
  const float* W_heads = (const float*)d_in[13];
  const float* a_heads = (const float*)d_in[14];
  const float* W_out   = (const float*)d_in[15];
  const float* a_out   = (const float*)d_in[16];
  const float* W_gcn   = (const float*)d_in[17];
  const float* a_fuse  = (const float*)d_in[18];
  const float* W_cls   = (const float*)d_in[19];
  const float* b_cls   = (const float*)d_in[20];

  // ---- workspace carve (~300 MB assumed available) ----
  size_t off = 0;
  char* base = (char*)d_ws;
  auto carve = [&](size_t bytes) -> char* {
    char* p = base + off;
    off += (bytes + 255) & ~(size_t)255;
    return p;
  };
  unsigned short* wAttT   = (unsigned short*)carve((size_t)EMB * EMB * 2);
  unsigned short* wHeadsT = (unsigned short*)carve((size_t)NHEAD * EMB * EMB * 2);
  unsigned short* wOutT   = (unsigned short*)carve((size_t)NHEAD * EMB * EMB * 2);
  unsigned short* wGcnT   = (unsigned short*)carve((size_t)EMB * EMB * 2);
  unsigned short* hb0     = (unsigned short*)carve((size_t)N0C * EMB * 2);
  unsigned short* hb1     = (unsigned short*)carve((size_t)N1C * EMB * 2);
  float*          Ebuf    = (float*)carve((size_t)N1C * N1C * 4);
  unsigned short* mixb    = (unsigned short*)carve((size_t)N1C * N1C * 2);
  float*          wh      = (float*)carve((size_t)N1C * EMB * 4);
  unsigned short* wht     = (unsigned short*)carve((size_t)EMB * N1C * 2);
  float*          xbuf    = (float*)carve((size_t)N1C * NHEAD * EMB * 4);
  unsigned short* xb      = (unsigned short*)carve((size_t)N1C * NHEAD * EMB * 2);
  float*          e1      = (float*)carve((size_t)N1C * 4);
  float*          e2      = (float*)carve((size_t)N1C * 4);
  float*          Lbuf    = (float*)carve((size_t)N1C * EMB * 4);
  float*          Zbuf    = (float*)carve((size_t)N1C * EMB * 4);
  float*          Fb0     = (float*)carve((size_t)N1C * EMB * 4);
  float*          Fb1     = (float*)carve((size_t)N1C * EMB * 4);
  float*          Fb2     = (float*)carve((size_t)N1C * EMB * 4);
  float*          Fb3     = (float*)carve((size_t)N1C * EMB * 4);
  float*          Rbuf    = (float*)carve((size_t)N0C * EMB * 4);
  float*          logits  = (float*)carve((size_t)BATCH * NCLS * 4);

  // ---- one-time weight transposes + input bf16 conversion ----
  unsigned int gW  = (unsigned int)(((size_t)EMB * EMB + 255) / 256);
  unsigned int gWo = (unsigned int)(((size_t)NHEAD * EMB * EMB + 255) / 256);
  tconv_bf16_kernel<<<gW, 256, 0, stream>>>(W_att, wAttT, EMB, EMB);
  for (int k = 0; k < NHEAD; ++k)
    tconv_bf16_kernel<<<gW, 256, 0, stream>>>(W_heads + (size_t)k * EMB * EMB,
                                              wHeadsT + (size_t)k * EMB * EMB, EMB, EMB);
  tconv_bf16_kernel<<<gWo, 256, 0, stream>>>(W_out, wOutT, NHEAD * EMB, EMB);
  tconv_bf16_kernel<<<gW, 256, 0, stream>>>(W_gcn, wGcnT, EMB, EMB);
  conv_bf16_kernel<<<(unsigned)(((size_t)N0C * EMB + 255) / 256), 256, 0, stream>>>(H,  hb0, (size_t)N0C * EMB);
  conv_bf16_kernel<<<(unsigned)(((size_t)N1C * EMB + 255) / 256), 256, 0, stream>>>(H_, hb1, (size_t)N1C * EMB);

  // ---- four branches (sequential on stream; big scratch reused) ----
  run_branch(stream, N0C, hb0, A0, Dg,  wAttT, wHeadsT, wOutT, wGcnT, a_att, a_heads, a_out,
             wh, wht, Ebuf, mixb, e1, e2, xbuf, xb, Lbuf, Zbuf, Fb0);
  run_branch(stream, N1C, hb1, A1, Dg_, wAttT, wHeadsT, wOutT, wGcnT, a_att, a_heads, a_out,
             wh, wht, Ebuf, mixb, e1, e2, xbuf, xb, Lbuf, Zbuf, Fb1);
  run_branch(stream, N1C, hb1, A2, Dg_, wAttT, wHeadsT, wOutT, wGcnT, a_att, a_heads, a_out,
             wh, wht, Ebuf, mixb, e1, e2, xbuf, xb, Lbuf, Zbuf, Fb2);
  run_branch(stream, N1C, hb1, A3, Dg_, wAttT, wHeadsT, wOutT, wGcnT, a_att, a_heads, a_out,
             wh, wht, Ebuf, mixb, e1, e2, xbuf, xb, Lbuf, Zbuf, Fb3);

  // ---- adaptive fusion over first N0 rows ----
  fuse_gate_kernel<<<N0C, 256, 0, stream>>>(Fb0, Fb1, Fb2, Fb3, a_fuse, Rbuf);

  // ---- fused gather + pos-enc + mask + classifier + softmax ----
  classify_kernel<<<BATCH * NCLS, 256, 0, stream>>>(Rbuf, doc_idx, doc_len, subp,
                                                    W_cls, b_cls, logits);
  softmax20_kernel<<<1, 32, 0, stream>>>(logits, (float*)d_out);
}
